// BerryPhaseCrossAttenuator_56126632624472
// MI455X (gfx1250) — compile-verified, matrix operand order checked by reading, not run on hardware
//
#include <hip/hip_runtime.h>
#include <hip/hip_bf16.h>

// ---------------------------------------------------------------------------
// BerryPhaseCrossAttenuator for MI455X (gfx1250, wave32, WMMA bf16 16x16x32)
//
// interference.mean == (1/H) * sum_h <vhat_h, that_h>^2
//                   == (1/H) * <pairfeat(v), pairfeat(t)>   (pairfeat in R^1024)
// => scores are one dense bf16 GEMM with K=1024.
// All GEMM operands are pre-converted to bf16 (A row-major, B^T row-major)
// so the WMMA K-loops are pure global_load_b128 + v_wmma.
// ---------------------------------------------------------------------------

typedef __attribute__((ext_vector_type(16))) __bf16 v16bf;
typedef __attribute__((ext_vector_type(8)))  float  v8f;

#define B_  2
#define N_  512
#define M_  512
#define D_  256
#define H_  64
#define KP_ 1024   // pair-feature depth

// ---- f32 -> bf16 (packed when hardware cvt builtin is available) ----------
__device__ __forceinline__ unsigned pack2bf(float lo, float hi) {
#if defined(__gfx1250__) && __has_builtin(__builtin_amdgcn_cvt_pk_bf16_f32)
    typedef __attribute__((ext_vector_type(2))) __bf16 v2bf;
    v2bf r = __builtin_amdgcn_cvt_pk_bf16_f32(lo, hi);
    return __builtin_bit_cast(unsigned, r);
#else
    unsigned short a = __builtin_bit_cast(unsigned short, (__bf16)lo);
    unsigned short b = __builtin_bit_cast(unsigned short, (__bf16)hi);
    return (unsigned)a | ((unsigned)b << 16);
#endif
}

__device__ __forceinline__ __bf16 f2bf(float f) {
    return (__bf16)f;
}

// ---- 16x32 bf16 fragment from row-major memory ----------------------------
// Element (idx, k) at base[idx*rowStride + k]; idx = lane&15.
// ISA 7.12.2 16-bit A layout: lanes 0-15 hold K {0..7,16..23},
// lanes 16-31 hold K {8..15,24..31}; pairs packed per dword.
__device__ __forceinline__ v16bf load_frag(const __bf16* __restrict__ base,
                                           int rowStride, int k0, int lane) {
    union { v16bf v; unsigned u[8]; } r;
    const __bf16* p = base + (lane & 15) * rowStride + k0 + (((lane >> 4) & 1) << 3);
#pragma unroll
    for (int v = 0; v < 8; ++v) {
        int kb = ((v & 4) << 2) + ((v & 3) << 1);   // 0,2,4,6,16,18,20,22
        r.u[v] = *(const unsigned*)(p + kb);
    }
    return r.v;
}

// ---- one 16x16 output tile: A (MxK row-major) x B^T (NxK row-major) -------
__device__ __forceinline__ v8f gemm_tile(const __bf16* __restrict__ A, int lda,
                                         const __bf16* __restrict__ Bt, int ldb,
                                         int K, int rowBase, int colBase, int lane) {
    v8f acc = {};
    const __bf16* a0 = A + (size_t)rowBase * lda;
    const __bf16* b0 = Bt + (size_t)colBase * ldb;
    for (int k0 = 0; k0 < K; k0 += 32) {
        v16bf a = load_frag(a0, lda, k0, lane);
        v16bf b = load_frag(b0, ldb, k0, lane);
        acc = __builtin_amdgcn_wmma_f32_16x16x32_bf16(false, a, false, b,
                                                      (short)0, acc, false, false);
    }
    return acc;
}

// ---------------------------------------------------------------------------
// P0a: elementwise f32 -> bf16 (n multiple of 8)
// ---------------------------------------------------------------------------
__global__ void cvt_bf16_kernel(const float* __restrict__ in,
                                __bf16* __restrict__ out, int n) {
    int i = (blockIdx.x * blockDim.x + threadIdx.x) * 8;
    if (i >= n) return;
    float4 f0 = *(const float4*)(in + i);
    float4 f1 = *(const float4*)(in + i + 4);
    uint4 o;
    o.x = pack2bf(f0.x, f0.y);
    o.y = pack2bf(f0.z, f0.w);
    o.z = pack2bf(f1.x, f1.y);
    o.w = pack2bf(f1.z, f1.w);
    *(uint4*)(out + i) = o;
}

// ---------------------------------------------------------------------------
// P0b: per-batch transpose + convert: in (B,R,Dd) f32 -> out (B,Dd,R) bf16
// ---------------------------------------------------------------------------
__global__ void cvtT_bf16_kernel(const float* __restrict__ in,
                                 __bf16* __restrict__ out, int R, int Dd) {
    int idx = blockIdx.x * blockDim.x + threadIdx.x;
    int half = R >> 1;
    int total = B_ * Dd * half;
    if (idx >= total) return;
    int rp = idx % half;
    int d  = (idx / half) % Dd;
    int b  = idx / (half * Dd);
    int r = rp * 2;
    float f0 = in[((size_t)b * R + r) * Dd + d];
    float f1 = in[((size_t)b * R + r + 1) * Dd + d];
    *(unsigned*)(out + ((size_t)b * Dd + d) * R + r) = pack2bf(f0, f1);
}

// ---------------------------------------------------------------------------
// K1: proj[r,n] = sum_k X[r,k]*W[n,k] + bias[n]   (X: R x D bf16, W: D x D bf16)
// ---------------------------------------------------------------------------
__global__ void berry_proj_kernel(const __bf16* __restrict__ X,
                                  const __bf16* __restrict__ Wb,
                                  const float* __restrict__ bias,
                                  float* __restrict__ out) {
    int lane = threadIdx.x & 31;
    int wave = threadIdx.x >> 5;
    int rowBase = blockIdx.x * 16;
    int colBase = blockIdx.y * 64 + wave * 16;

    v8f acc = gemm_tile(X, D_, Wb, D_, D_, rowBase, colBase, lane);

    int n  = colBase + (lane & 15);
    int hl = (lane >> 4) & 1;
    float bn = bias[n];
#pragma unroll
    for (int i = 0; i < 8; ++i)
        out[(size_t)(rowBase + i + 8 * hl) * D_ + n] = acc[i] + bn;
}

// ---------------------------------------------------------------------------
// K2: normalize head quaternions, emit 16 bf16 pair products per head.
// ---------------------------------------------------------------------------
__global__ void berry_pairs_kernel(const float* __restrict__ proj,
                                   __bf16* __restrict__ P, int total) {
    int idx = blockIdx.x * blockDim.x + threadIdx.x;
    if (idx >= total) return;
    int row  = idx >> 6;
    int head = idx & 63;
    float4 q = *(const float4*)(proj + (size_t)row * D_ + head * 4);
    float inv = 1.0f / (sqrtf(q.x * q.x + q.y * q.y + q.z * q.z + q.w * q.w) + 1e-8f);
    float v[4] = {q.x * inv, q.y * inv, q.z * inv, q.w * inv};
    unsigned* o = (unsigned*)(P + (size_t)row * KP_ + head * 16);
#pragma unroll
    for (int i = 0; i < 4; ++i) {
        o[2 * i]     = pack2bf(v[i] * v[0], v[i] * v[1]);
        o[2 * i + 1] = pack2bf(v[i] * v[2], v[i] * v[3]);
    }
}

// ---------------------------------------------------------------------------
// K3: S[b] = Vp[b] (N x KP) @ Tp[b]^T, bf16 WMMA K=1024, f32 out.
// ---------------------------------------------------------------------------
__global__ void berry_score_kernel(const __bf16* __restrict__ Vp,
                                   const __bf16* __restrict__ Tp,
                                   float* __restrict__ S) {
    int lane = threadIdx.x & 31;
    int wave = threadIdx.x >> 5;
    int b = blockIdx.z;
    int rowBase = blockIdx.x * 16;
    int colBase = blockIdx.y * 64 + wave * 16;

    v8f acc = gemm_tile(Vp + (size_t)b * N_ * KP_, KP_,
                        Tp + (size_t)b * M_ * KP_, KP_,
                        KP_, rowBase, colBase, lane);

    float* out = S + (size_t)b * N_ * M_;
    int n  = colBase + (lane & 15);
    int hl = (lane >> 4) & 1;
#pragma unroll
    for (int i = 0; i < 8; ++i)
        out[(size_t)(rowBase + i + 8 * hl) * M_ + n] = acc[i];
}

// ---------------------------------------------------------------------------
// K4: row softmax (scale 1/(H*sqrt(D)) = 1/1024); writes bf16 attn and attn^T.
// ---------------------------------------------------------------------------
__global__ void berry_softmax_kernel(const float* __restrict__ S,
                                     __bf16* __restrict__ Attn,
                                     __bf16* __restrict__ AttnT) {
    __shared__ float redmax[8];
    __shared__ float redsum[8];
    int row = blockIdx.x;               // global row in [0, B*N)
    int b = row >> 9, n = row & 511;
    const float* p = S + (size_t)row * M_;
    int t = threadIdx.x;                // 256 threads = 8 waves
    int lane = t & 31, wv = t >> 5;
    const float scale = 1.0f / 1024.0f;

    float x0 = p[t] * scale;
    float x1 = p[t + 256] * scale;

    float mx = fmaxf(x0, x1);
#pragma unroll
    for (int off = 16; off > 0; off >>= 1)
        mx = fmaxf(mx, __shfl_xor(mx, off, 32));
    if (lane == 0) redmax[wv] = mx;
    __syncthreads();
    if (wv == 0) {
        float m2 = (lane < 8) ? redmax[lane] : -3.4e38f;
#pragma unroll
        for (int off = 4; off > 0; off >>= 1)
            m2 = fmaxf(m2, __shfl_xor(m2, off, 32));
        if (lane == 0) redmax[0] = m2;
    }
    __syncthreads();
    mx = redmax[0];

    float e0 = __expf(x0 - mx);
    float e1 = __expf(x1 - mx);
    float s = e0 + e1;
#pragma unroll
    for (int off = 16; off > 0; off >>= 1)
        s += __shfl_xor(s, off, 32);
    if (lane == 0) redsum[wv] = s;
    __syncthreads();
    if (wv == 0) {
        float s2 = (lane < 8) ? redsum[lane] : 0.0f;
#pragma unroll
        for (int off = 4; off > 0; off >>= 1)
            s2 += __shfl_xor(s2, off, 32);
        if (lane == 0) redsum[0] = s2;
    }
    __syncthreads();
    float inv = 1.0f / redsum[0];
    float a0 = e0 * inv, a1 = e1 * inv;

    __bf16* ar = Attn + ((size_t)b * N_ + n) * M_;
    ar[t]       = f2bf(a0);
    ar[t + 256] = f2bf(a1);
    __bf16* at = AttnT + (size_t)b * M_ * N_;
    at[(size_t)t * N_ + n]         = f2bf(a0);
    at[(size_t)(t + 256) * N_ + n] = f2bf(a1);
}

// ---------------------------------------------------------------------------
// K5: out[bz,m,n] = res[bz,m,n] + h * sum_k A[m,k] * Bt[n,k]   (bf16 operands)
// ---------------------------------------------------------------------------
__global__ void berry_out_kernel(const __bf16* __restrict__ Abf, long aBatch, int lda,
                                 const __bf16* __restrict__ Btbf, long bBatch, int ldb,
                                 const float* __restrict__ Res,
                                 float* __restrict__ Out,
                                 const float* __restrict__ hPtr, int K) {
    int lane = threadIdx.x & 31;
    int wave = threadIdx.x >> 5;
    int bz = blockIdx.z;
    int rowBase = blockIdx.x * 16;
    int colBase = blockIdx.y * 64 + wave * 16;

    v8f acc = gemm_tile(Abf + (size_t)bz * aBatch, lda,
                        Btbf + (size_t)bz * bBatch, ldb,
                        K, rowBase, colBase, lane);

    float hh = hPtr[0];
    int n  = colBase + (lane & 15);
    int hl = (lane >> 4) & 1;
    size_t base = (size_t)bz * N_ * D_;
#pragma unroll
    for (int i = 0; i < 8; ++i) {
        size_t off = base + (size_t)(rowBase + i + 8 * hl) * D_ + n;
        Out[off] = Res[off] + hh * acc[i];
    }
}

// ---------------------------------------------------------------------------
extern "C" void kernel_launch(void* const* d_in, const int* in_sizes, int n_in,
                              void* d_out, int out_size, void* d_ws, size_t ws_size,
                              hipStream_t stream) {
    const float* vision = (const float*)d_in[0];
    const float* text   = (const float*)d_in[1];
    const float* Wv     = (const float*)d_in[2];
    const float* bv     = (const float*)d_in[3];
    const float* Wt     = (const float*)d_in[4];
    const float* bt     = (const float*)d_in[5];
    const float* hPtr   = (const float*)d_in[6];

    float* outv = (float*)d_out;                       // (B, N, D)
    float* outt = outv + (size_t)B_ * N_ * D_;         // (B, M, D)

    const size_t FEAT = (size_t)B_ * N_ * D_;          // 262144
    const size_t WSZ  = (size_t)D_ * D_;               // 65536
    const size_t ATT  = (size_t)B_ * N_ * M_;          // 524288
    const size_t PFT  = (size_t)B_ * N_ * KP_;         // 1048576

    // workspace layout (~12.5 MB)
    float*  vproj    = (float*)d_ws;                   // FEAT f32
    float*  tproj    = vproj + FEAT;                   // FEAT f32
    float*  S        = tproj + FEAT;                   // ATT  f32
    __bf16* vision_b = (__bf16*)(S + ATT);             // FEAT
    __bf16* text_b   = vision_b + FEAT;                // FEAT
    __bf16* visionT  = text_b + FEAT;                  // FEAT (B, D, N)
    __bf16* textT    = visionT + FEAT;                 // FEAT (B, D, M)
    __bf16* Wv_b     = textT + FEAT;                   // WSZ
    __bf16* Wt_b     = Wv_b + WSZ;                     // WSZ
    __bf16* Vp       = Wt_b + WSZ;                     // PFT
    __bf16* Tp       = Vp + PFT;                       // PFT
    __bf16* Attn     = Tp + PFT;                       // ATT
    __bf16* AttnT    = Attn + ATT;                     // ATT

    // P0: one-shot bf16 conversions (each element converted exactly once)
    cvt_bf16_kernel<<<(int)(FEAT / 8 + 255) / 256, 256, 0, stream>>>(vision, vision_b, (int)FEAT);
    cvt_bf16_kernel<<<(int)(FEAT / 8 + 255) / 256, 256, 0, stream>>>(text,   text_b,   (int)FEAT);
    cvt_bf16_kernel<<<(int)(WSZ / 8 + 255) / 256, 256, 0, stream>>>(Wv, Wv_b, (int)WSZ);
    cvt_bf16_kernel<<<(int)(WSZ / 8 + 255) / 256, 256, 0, stream>>>(Wt, Wt_b, (int)WSZ);
    int tT = B_ * D_ * (N_ / 2);
    cvtT_bf16_kernel<<<(tT + 255) / 256, 256, 0, stream>>>(vision, visionT, N_, D_);
    cvtT_bf16_kernel<<<(tT + 255) / 256, 256, 0, stream>>>(text,   textT,   M_, D_);

    // K1: projections (rows flattened over batch)
    berry_proj_kernel<<<dim3((B_ * N_) / 16, D_ / 64), 128, 0, stream>>>(vision_b, Wv_b, bv, vproj);
    berry_proj_kernel<<<dim3((B_ * M_) / 16, D_ / 64), 128, 0, stream>>>(text_b,   Wt_b, bt, tproj);

    // K2: quaternion-normalized pair features
    int totalv = B_ * N_ * H_;
    berry_pairs_kernel<<<(totalv + 255) / 256, 256, 0, stream>>>(vproj, Vp, totalv);
    berry_pairs_kernel<<<(totalv + 255) / 256, 256, 0, stream>>>(tproj, Tp, totalv);

    // K3: interference scores, one bf16 GEMM per batch, K = 1024
    berry_score_kernel<<<dim3(N_ / 16, M_ / 64, B_), 128, 0, stream>>>(Vp, Tp, S);

    // K4: softmax rows -> bf16 attn and attn^T
    berry_softmax_kernel<<<B_ * N_, 256, 0, stream>>>(S, Attn, AttnT);

    // K5: residual cross-attention outputs
    // out_vision = vision + h * attn @ text        (Bt = text^T, per-batch)
    berry_out_kernel<<<dim3(N_ / 16, D_ / 64, B_), 128, 0, stream>>>(
        Attn, (long)N_ * M_, M_,
        textT, (long)D_ * M_, M_,
        vision, outv, hPtr, M_);
    // out_text = text + h * attn^T @ vision        (Bt = vision^T, per-batch)
    berry_out_kernel<<<dim3(M_ / 16, D_ / 64, B_), 128, 0, stream>>>(
        AttnT, (long)M_ * N_, N_,
        visionT, (long)D_ * N_, N_,
        text, outt, hPtr, N_);
}
